// Attention_48464410968430
// MI455X (gfx1250) — matrix operand order repaired and checked by reading, not compiled, not run on hardware
//
#include <hip/hip_runtime.h>
#include <hip/hip_bf16.h>

// ---------------------------------------------------------------------------
// CDNA5 (gfx1250) causal multi-head attention, bf16 WMMA pipeline.
//   B=4, S=2048, D=1024, H=16, HD=64
// Stages: cvt x->bf16 | cvt+transpose weights->bf16 | QKV GEMM (wmma, async
//         global->LDS double-buffered staging) | flash attention (wmma QK^T +
//         PV, online softmax) | out-proj GEMM (same async staging).
// Workspace layout (72 MB total, offsets in MB):
//   [ 0..16)  xbf   [8192][1024] bf16   (reused as attn output after stage 4)
//   [16..22)  waT   [3072][1024] bf16   (w_attn transposed, N-major)
//   [22..24)  wpT   [1024][1024] bf16   (w_proj transposed)
//   [24..40)  Q     [64][2048][64] bf16 (bh-major)
//   [40..56)  K     [64][2048][64] bf16
//   [56..72)  vT    [64][64][2048] bf16 (V transposed: hd-major, s contiguous)
// ---------------------------------------------------------------------------

typedef __attribute__((ext_vector_type(16))) __bf16 v16bf;
typedef __attribute__((ext_vector_type(8)))  float  v8f;
typedef __attribute__((ext_vector_type(4)))  float  v4f;

#define NEG_BIG (-1e30f)

union FragBF {
  v16bf v;
  v4f   f4[2];
};

__device__ __forceinline__ unsigned short f2bf(float f) {
  union { float f; unsigned u; } x;
  x.f = f;
  unsigned u = x.u;
  u += 0x7FFFu + ((u >> 16) & 1u);   // round-to-nearest-even
  return (unsigned short)(u >> 16);
}

// One 16-byte async global->LDS copy per lane (ASYNCcnt-tracked DMA path).
// lds_off: low 32 bits of the flat shared-memory address (= LDS allocation
// offset per ISA 10.2 aperture mapping).
__device__ __forceinline__ void issue_async_16B(unsigned lds_off, const void* gaddr) {
  asm volatile("global_load_async_to_lds_b128 %0, %1, off"
               :: "v"(lds_off), "v"(gaddr) : "memory");
}

// ---------------------------------------------------------------- stage 1 ---
__global__ void cvt_f32_bf16(const float* __restrict__ in,
                             unsigned short* __restrict__ out, int n) {
  int i = blockIdx.x * blockDim.x + threadIdx.x;
  if (i < n) out[i] = f2bf(in[i]);
}

// in: [K][N] f32 -> out: [N][K] bf16
__global__ void transpose_cvt(const float* __restrict__ in,
                              unsigned short* __restrict__ out, int K, int N) {
  int i = blockIdx.x * blockDim.x + threadIdx.x;
  if (i < K * N) {
    int n = i / K;
    int k = i - n * K;
    out[i] = f2bf(in[(size_t)k * N + n]);
  }
}

// ---------------------------------------------------------------- stage 2 ---
// C[8192][3072] = xbf[8192][1024] @ w_attn[1024][3072] + b_attn
// Block: 128(M) x 128(N); 8 waves in 2(M) x 4(N); wave tile 64x32 = 4x2 wmma.
// A/B tiles staged to LDS with double-buffered async copies:
//   issue(next) -> s_wait_asynccnt 4 -> barrier -> 8 wmma -> barrier.
__global__ __launch_bounds__(256) void qkv_gemm(
    const unsigned short* __restrict__ xbf,
    const unsigned short* __restrict__ waT,
    const float* __restrict__ b_attn,
    unsigned short* __restrict__ qbuf,
    unsigned short* __restrict__ kbuf,
    unsigned short* __restrict__ vT) {
  __shared__ unsigned short Asm[2][128][32];   // [buf][row][k]  8KB each
  __shared__ unsigned short Bsm[2][128][32];   // [buf][col][k]  8KB each

  const int lane  = threadIdx.x & 31;
  const int wid   = threadIdx.x >> 5;
  const int hi    = lane >> 4;
  const int laneM = lane & 15;
  const int laneN = lane & 15;
  const int kA = hi * 8;    // A frag: hi lanes hold K 8..15 / 24..31
  const int kB = hi * 8;    // B frag: hi lanes hold K 8..15 / 24..31
  const int m0blk = blockIdx.y * 128;
  const int n0blk = blockIdx.x * 128;
  const int rowA0 = (wid & 1) * 64;
  const int colB0 = (wid >> 1) * 32;

  // 512 16B chunks per tile, 256 threads -> 2 async ops per thread per tile.
#define ISSUE_TILES(buf, k0)                                                   \
  {                                                                            \
    _Pragma("unroll")                                                          \
    for (int c = 0; c < 2; ++c) {                                              \
      const int chunk = (int)threadIdx.x + c * 256;                            \
      const int rr = chunk >> 2, sub = chunk & 3;                              \
      issue_async_16B((unsigned)(uintptr_t)&Asm[buf][rr][sub * 8],             \
                      xbf + (size_t)(m0blk + rr) * 1024 + (k0) + sub * 8);     \
      issue_async_16B((unsigned)(uintptr_t)&Bsm[buf][rr][sub * 8],             \
                      waT + (size_t)(n0blk + rr) * 1024 + (k0) + sub * 8);     \
    }                                                                          \
  }

  v8f acc[4][2] = {};

  ISSUE_TILES(0, 0);
  for (int t = 0; t < 32; ++t) {
    const int cur = t & 1;
    if (t < 31) {
      ISSUE_TILES(cur ^ 1, (t + 1) * 32);
      asm volatile("s_wait_asynccnt 0x4" ::: "memory");  // current buf landed
    } else {
      asm volatile("s_wait_asynccnt 0x0" ::: "memory");
    }
    __syncthreads();

    FragBF a[4];
#pragma unroll
    for (int i = 0; i < 4; i++) {
      const unsigned short* p = &Asm[cur][rowA0 + i * 16 + laneM][kA];
      a[i].f4[0] = *(const v4f*)(p);
      a[i].f4[1] = *(const v4f*)(p + 16);
    }
    FragBF bm[2];
#pragma unroll
    for (int j = 0; j < 2; j++) {
      const unsigned short* p = &Bsm[cur][colB0 + j * 16 + laneN][kB];
      bm[j].f4[0] = *(const v4f*)(p);
      bm[j].f4[1] = *(const v4f*)(p + 16);
    }
#pragma unroll
    for (int i = 0; i < 4; i++)
#pragma unroll
      for (int j = 0; j < 2; j++)
        acc[i][j] = __builtin_amdgcn_wmma_f32_16x16x32_bf16(
            false, a[i].v, false, bm[j].v, (short)0, acc[i][j], false, false);
    __syncthreads();
  }

  // Epilogue: +bias, split q/k/v into head-major layouts (V transposed).
#pragma unroll
  for (int i = 0; i < 4; i++) {
#pragma unroll
    for (int j = 0; j < 2; j++) {
      const int n      = n0blk + colB0 + j * 16 + laneN;
      const float bias = b_attn[n];
      const int which = n >> 10;   // 0=q 1=k 2=v
      const int cc    = n & 1023;
      const int h     = cc >> 6;
      const int hd    = cc & 63;
#pragma unroll
      for (int r = 0; r < 8; r++) {
        const int row = m0blk + rowA0 + i * 16 + r + hi * 8;
        const int b_  = row >> 11;
        const int s   = row & 2047;
        const int bh  = b_ * 16 + h;
        const unsigned short bv = f2bf(acc[i][j][r] + bias);
        if (which == 0)      qbuf[((size_t)bh * 2048 + s) * 64 + hd] = bv;
        else if (which == 1) kbuf[((size_t)bh * 2048 + s) * 64 + hd] = bv;
        else                 vT[((size_t)bh * 64 + hd) * 2048 + s]   = bv;
      }
    }
  }
#undef ISSUE_TILES
}

// ---------------------------------------------------------------- stage 3 ---
// Flash attention: block = (128 queries x one bh); wave owns 16 queries.
// Online softmax over 64-key chunks; causal trip count per wave (no barriers).
__global__ __launch_bounds__(256) void flash_attn(
    const unsigned short* __restrict__ qbuf,
    const unsigned short* __restrict__ kbuf,
    const unsigned short* __restrict__ vT,
    unsigned short* __restrict__ attnbf) {
  __shared__ unsigned short Plds[8][16][72];   // wave-private P tiles (padded)
  const int lane  = threadIdx.x & 31;
  const int wid   = threadIdx.x >> 5;
  const int hi    = lane >> 4;
  const int laneM = lane & 15;
  const int laneN = lane & 15;
  const int kA = hi * 8;
  const int kB = hi * 8;
  const int bh    = blockIdx.y;
  const int qBase = blockIdx.x * 128 + wid * 16;

  // Q A-fragments (held for whole kernel): 16 rows x 64 hd -> 2 K=32 frags.
  FragBF qa[2];
  {
    const unsigned short* p = qbuf + ((size_t)bh * 2048 + qBase + laneM) * 64;
#pragma unroll
    for (int c = 0; c < 2; c++) {
      qa[c].f4[0] = *(const v4f*)(p + c * 32 + kA);
      qa[c].f4[1] = *(const v4f*)(p + c * 32 + kA + 16);
    }
  }

  float mrow[8], lrow[8], alph[8];
#pragma unroll
  for (int r = 0; r < 8; r++) { mrow[r] = NEG_BIG; lrow[r] = 0.0f; }
  v8f O[4] = {};

  for (int kv0 = 0; kv0 < qBase + 16; kv0 += 64) {
    // ---- scores: 4 x (16q x 16k) tiles, contraction over hd=64 (2 wmma) ----
    v8f s[4] = {};
#pragma unroll
    for (int j = 0; j < 4; j++) {
#pragma unroll
      for (int c = 0; c < 2; c++) {
        FragBF kb;
        const unsigned short* p =
            kbuf + ((size_t)bh * 2048 + kv0 + j * 16 + laneN) * 64 + c * 32 + kB;
        kb.f4[0] = *(const v4f*)(p);
        kb.f4[1] = *(const v4f*)(p + 16);
        s[j] = __builtin_amdgcn_wmma_f32_16x16x32_bf16(
            false, qa[c].v, false, kb.v, (short)0, s[j], false, false);
      }
    }
    // ---- scale + causal mask (C layout: row = r + hi*8, col = lane&15) ----
#pragma unroll
    for (int j = 0; j < 4; j++) {
      const int key = kv0 + j * 16 + laneN;
#pragma unroll
      for (int r = 0; r < 8; r++) {
        const int qi = qBase + r + hi * 8;
        const float v = s[j][r] * 0.125f;   // 1/sqrt(64)
        s[j][r] = (key > qi) ? NEG_BIG : v;
      }
    }
    // ---- online softmax update ----
#pragma unroll
    for (int r = 0; r < 8; r++) {
      float mx = fmaxf(fmaxf(s[0][r], s[1][r]), fmaxf(s[2][r], s[3][r]));
#pragma unroll
      for (int off = 1; off < 16; off <<= 1)
        mx = fmaxf(mx, __shfl_xor(mx, off, 32));
      const float mnew = fmaxf(mrow[r], mx);
      alph[r] = __expf(mrow[r] - mnew);
      mrow[r] = mnew;
      float rs = 0.0f;
#pragma unroll
      for (int j = 0; j < 4; j++) {
        const float pv = __expf(s[j][r] - mnew);
        s[j][r] = pv;
        rs += pv;
      }
#pragma unroll
      for (int off = 1; off < 16; off <<= 1)
        rs += __shfl_xor(rs, off, 32);
      lrow[r] = lrow[r] * alph[r] + rs;
    }
#pragma unroll
    for (int t = 0; t < 4; t++)
#pragma unroll
      for (int r = 0; r < 8; r++)
        O[t][r] *= alph[r];
    // ---- P: C layout -> A layout via wave-private LDS tile ----
#pragma unroll
    for (int j = 0; j < 4; j++)
#pragma unroll
      for (int r = 0; r < 8; r++)
        Plds[wid][r + hi * 8][j * 16 + laneN] = f2bf(s[j][r]);
    asm volatile("s_wait_dscnt 0" ::: "memory");
    FragBF pa[2];
#pragma unroll
    for (int c = 0; c < 2; c++) {
      const unsigned short* p = &Plds[wid][laneM][c * 32 + kA];
      pa[c].f4[0] = *(const v4f*)(p);
      pa[c].f4[1] = *(const v4f*)(p + 16);
    }
    // ---- O += P @ V (vT gives contiguous per-column loads) ----
#pragma unroll
    for (int t = 0; t < 4; t++) {
#pragma unroll
      for (int c = 0; c < 2; c++) {
        FragBF vb;
        const unsigned short* p =
            vT + ((size_t)bh * 64 + t * 16 + laneN) * 2048 + kv0 + c * 32 + kB;
        vb.f4[0] = *(const v4f*)(p);
        vb.f4[1] = *(const v4f*)(p + 16);
        O[t] = __builtin_amdgcn_wmma_f32_16x16x32_bf16(
            false, pa[c].v, false, vb.v, (short)0, O[t], false, false);
      }
    }
  }

  // ---- normalize + merge heads: attnbf[b][s][h*64+hd] ----
  const int b_ = bh >> 4, h = bh & 15;
#pragma unroll
  for (int t = 0; t < 4; t++) {
#pragma unroll
    for (int r = 0; r < 8; r++) {
      const int row = qBase + r + hi * 8;
      const float val = O[t][r] / lrow[r];
      attnbf[((size_t)b_ * 2048 + row) * 1024 + h * 64 + t * 16 + laneN] = f2bf(val);
    }
  }
}

// ---------------------------------------------------------------- stage 4 ---
// out[8192][1024] = attnbf @ w_proj + b_proj  (f32 output), same async staging.
__global__ __launch_bounds__(256) void proj_gemm(
    const unsigned short* __restrict__ abf,
    const unsigned short* __restrict__ wpT,
    const float* __restrict__ b_proj,
    float* __restrict__ out) {
  __shared__ unsigned short Asm[2][128][32];
  __shared__ unsigned short Bsm[2][128][32];

  const int lane  = threadIdx.x & 31;
  const int wid   = threadIdx.x >> 5;
  const int hi    = lane >> 4;
  const int laneM = lane & 15;
  const int laneN = lane & 15;
  const int kA = hi * 8;
  const int kB = hi * 8;
  const int m0blk = blockIdx.y * 128;
  const int n0blk = blockIdx.x * 128;
  const int rowA0 = (wid & 1) * 64;
  const int colB0 = (wid >> 1) * 32;

#define ISSUE_TILES(buf, k0)                                                   \
  {                                                                            \
    _Pragma("unroll")                                                          \
    for (int c = 0; c < 2; ++c) {                                              \
      const int chunk = (int)threadIdx.x + c * 256;                            \
      const int rr = chunk >> 2, sub = chunk & 3;                              \
      issue_async_16B((unsigned)(uintptr_t)&Asm[buf][rr][sub * 8],             \
                      abf + (size_t)(m0blk + rr) * 1024 + (k0) + sub * 8);     \
      issue_async_16B((unsigned)(uintptr_t)&Bsm[buf][rr][sub * 8],             \
                      wpT + (size_t)(n0blk + rr) * 1024 + (k0) + sub * 8);     \
    }                                                                          \
  }

  v8f acc[4][2] = {};

  ISSUE_TILES(0, 0);
  for (int t = 0; t < 32; ++t) {
    const int cur = t & 1;
    if (t < 31) {
      ISSUE_TILES(cur ^ 1, (t + 1) * 32);
      asm volatile("s_wait_asynccnt 0x4" ::: "memory");
    } else {
      asm volatile("s_wait_asynccnt 0x0" ::: "memory");
    }
    __syncthreads();

    FragBF a[4];
#pragma unroll
    for (int i = 0; i < 4; i++) {
      const unsigned short* p = &Asm[cur][rowA0 + i * 16 + laneM][kA];
      a[i].f4[0] = *(const v4f*)(p);
      a[i].f4[1] = *(const v4f*)(p + 16);
    }
    FragBF bm[2];
#pragma unroll
    for (int j = 0; j < 2; j++) {
      const unsigned short* p = &Bsm[cur][colB0 + j * 16 + laneN][kB];
      bm[j].f4[0] = *(const v4f*)(p);
      bm[j].f4[1] = *(const v4f*)(p + 16);
    }
#pragma unroll
    for (int i = 0; i < 4; i++)
#pragma unroll
      for (int j = 0; j < 2; j++)
        acc[i][j] = __builtin_amdgcn_wmma_f32_16x16x32_bf16(
            false, a[i].v, false, bm[j].v, (short)0, acc[i][j], false, false);
    __syncthreads();
  }

#pragma unroll
  for (int i = 0; i < 4; i++) {
#pragma unroll
    for (int j = 0; j < 2; j++) {
      const int n = n0blk + colB0 + j * 16 + laneN;
      const float bias = b_proj[n];
#pragma unroll
      for (int r = 0; r < 8; r++) {
        const int row = m0blk + rowA0 + i * 16 + r + hi * 8;
        out[(size_t)row * 1024 + n] = acc[i][j][r] + bias;
      }
    }
  }
#undef ISSUE_TILES
}

// --------------------------------------------------------------------------
extern "C" void kernel_launch(void* const* d_in, const int* in_sizes, int n_in,
                              void* d_out, int out_size, void* d_ws, size_t ws_size,
                              hipStream_t stream) {
  (void)in_sizes; (void)n_in; (void)out_size; (void)ws_size;
  const float* x      = (const float*)d_in[0];
  // d_in[1] = boolean mask: strictly causal, handled analytically in-kernel.
  const float* w_attn = (const float*)d_in[2];
  const float* b_attn = (const float*)d_in[3];
  const float* w_proj = (const float*)d_in[4];
  const float* b_proj = (const float*)d_in[5];
  float* out = (float*)d_out;

  char* ws = (char*)d_ws;
  unsigned short* xbf = (unsigned short*)(ws);                           // 16 MB
  unsigned short* waT = (unsigned short*)(ws + (size_t)16 * (1u << 20)); // 6 MB
  unsigned short* wpT = (unsigned short*)(ws + (size_t)22 * (1u << 20)); // 2 MB
  unsigned short* qb  = (unsigned short*)(ws + (size_t)24 * (1u << 20)); // 16 MB
  unsigned short* kb  = (unsigned short*)(ws + (size_t)40 * (1u << 20)); // 16 MB
  unsigned short* vT  = (unsigned short*)(ws + (size_t)56 * (1u << 20)); // 16 MB

  const int NX = 8192 * 1024;
  cvt_f32_bf16<<<(NX + 255) / 256, 256, 0, stream>>>(x, xbf, NX);
  const int NWA = 1024 * 3072;
  transpose_cvt<<<(NWA + 255) / 256, 256, 0, stream>>>(w_attn, waT, 1024, 3072);
  const int NWP = 1024 * 1024;
  transpose_cvt<<<(NWP + 255) / 256, 256, 0, stream>>>(w_proj, wpT, 1024, 1024);

  qkv_gemm<<<dim3(24, 64), 256, 0, stream>>>(xbf, waT, b_attn, qb, kb, vT);
  flash_attn<<<dim3(16, 64), 256, 0, stream>>>(qb, kb, vT, xbf /*reuse*/);
  proj_gemm<<<dim3(8, 64), 256, 0, stream>>>(xbf, wpT, b_proj, out);
}
